// GeometryGNN_54657753809377
// MI455X (gfx1250) — matrix operand order repaired and checked by reading, not compiled, not run on hardware
//
#include <hip/hip_runtime.h>
#include <hip/hip_bf16.h>
#include <stdint.h>

// ---------------------------------------------------------------------------
// GeometryGNN on MI455X (gfx1250): triangle-GIN collapses to a chain of 6
// dense GEMMs [B,H]x[H,H]. bf16 WMMA (v_wmma_f32_16x16x32_bf16); A and
// (pre-transposed) B tiles staged by the Tensor Data Mover with hardware
// LDS row padding; TENSORcnt-paced LDS double buffer; fragments load as
// two ds_load_b128 each. Weights converted+transposed to bf16 once.
// ---------------------------------------------------------------------------

typedef __bf16 bf16;
typedef __attribute__((ext_vector_type(16))) __bf16   v16bf;
typedef __attribute__((ext_vector_type(8)))  float    v8f;
typedef __attribute__((ext_vector_type(4)))  uint32_t u32x4;
typedef __attribute__((ext_vector_type(8)))  uint32_t u32x8;

static __device__ __forceinline__ bf16 f2bf(float f) {
    union { float f; uint32_t u; } v; v.f = f;
    uint32_t r = v.u + 0x7FFFu + ((v.u >> 16) & 1u);   // round-to-nearest-even
    return __builtin_bit_cast(bf16, (uint16_t)(r >> 16));
}
static __device__ __forceinline__ float bf2f(bf16 b) {
    uint32_t u = ((uint32_t)__builtin_bit_cast(uint16_t, b)) << 16;
    return __builtin_bit_cast(float, u);
}

// ---------------------------------------------------------------------------
// TDM: one 2-D tile DMA, global -> LDS, hardware LDS row padding.
// D# per CDNA5 ISA. pad_interval p pads after
// 2^(p+1) dwords; pad_amount a adds (a+1) dwords of pad.
// ---------------------------------------------------------------------------
static __device__ __forceinline__ void tdm_load_2d(
    uint32_t lds_off, const void* gptr,
    uint32_t tensor_d0, uint32_t tensor_d1,
    uint32_t tile_d0, uint32_t tile_d1,
    uint64_t stride0_elems, uint32_t pad_interval, uint32_t pad_amount)
{
    const uint64_t ga = (uint64_t)(uintptr_t)gptr;
    u32x4 g0;
    g0[0] = 1u;                                   // count=1, user descriptor
    g0[1] = lds_off;                              // lds_addr (bytes)
    g0[2] = (uint32_t)ga;                         // global_addr[31:0]
    g0[3] = (uint32_t)((ga >> 32) & 0x01FFFFFFu)  // global_addr[56:32]
          | (2u << 30);                           // type = 2 ("image")
    u32x8 g1;
    g1[0] = (1u << 16)                            // data_size = 2 bytes
          | (1u << 20)                            // pad_enable
          | (pad_interval << 22)
          | (pad_amount << 25);
    g1[1] = (tensor_d0 & 0xFFFFu) << 16;          // tensor_dim0[15:0]
    g1[2] = (tensor_d0 >> 16) | ((tensor_d1 & 0xFFFFu) << 16);
    g1[3] = (tensor_d1 >> 16) | (tile_d0 << 16);  // tile_dim0
    g1[4] = tile_d1 & 0xFFFFu;                    // tile_dim1 (tile_dim2=0)
    g1[5] = (uint32_t)(stride0_elems & 0xFFFFFFFFu);
    g1[6] = (uint32_t)((stride0_elems >> 32) & 0xFFFFu);
    g1[7] = 0u;
    const u32x4 gz = {0u, 0u, 0u, 0u};
    asm volatile("tensor_load_to_lds %0, %1, %2, %3"
                 :: "s"(g0), "s"(g1), "s"(gz), "s"(gz)
                 : "memory");
}

// ---------------------------------------------------------------------------
// Tiled WMMA GEMM:  C[M,N](bf16) = scale * relu(A @ Bt^T + bias) + extra
// A[M,K] bf16 row-major; Bt[N,K] bf16 (weights pre-transposed, K-contiguous).
// 256 threads = 8 waves (4M x 2N), BM=BN=128, BK=32; each wave owns 2x4
// 16x16 accumulators. TDM double buffer: issue tile t+1, wait tensorcnt<=2.
// LDS row stride 40 bf16 (80 B, 16B-aligned) via TDM pad => fragments are
// two ds_load_b128 each, conflict-free (20-dword lane stride).
// ---------------------------------------------------------------------------
#define BM 128
#define BN 128
#define BK 32
#define LDT 40    // LDS row stride in bf16: 32 data + 8 pad (4 dwords)

__global__ __launch_bounds__(256)
void gemm_bf16_wmma(const bf16* __restrict__ A, const bf16* __restrict__ Bt,
                    const float* __restrict__ bias, const float* __restrict__ extra,
                    float scale, bf16* __restrict__ C, int M, int N, int K)
{
    __shared__ __align__(16) bf16 As[2][BM * LDT];   // [row][k] padded
    __shared__ __align__(16) bf16 Bs[2][BN * LDT];   // [col][k] padded

    const int tid  = threadIdx.x;
    const int lane = tid & 31;
    const int wave = tid >> 5;
    const int wm   = wave >> 1;        // 0..3 (M)
    const int wn   = wave & 1;         // 0..1 (N)
    const int lrow = lane & 15;
    const int lhi  = lane >> 4;        // 0/1

    const int rowTile = blockIdx.y * BM;
    const int colTile = blockIdx.x * BN;

    v8f acc[2][4];
    #pragma unroll
    for (int mi = 0; mi < 2; ++mi)
        #pragma unroll
        for (int ni = 0; ni < 4; ++ni)
            acc[mi][ni] = v8f{};

    const bf16* Atile = A  + (size_t)rowTile * K;
    const bf16* Btile = Bt + (size_t)colTile * K;
    const int T = K / BK;

    if (wave == 0) {
        tdm_load_2d((uint32_t)(uintptr_t)&As[0][0], Atile,
                    (uint32_t)K, (uint32_t)M, BK, BM, (uint64_t)K, 3u, 3u);
        tdm_load_2d((uint32_t)(uintptr_t)&Bs[0][0], Btile,
                    (uint32_t)K, (uint32_t)N, BK, BN, (uint64_t)K, 3u, 3u);
    }

    for (int t = 0; t < T; ++t) {
        const int cur = t & 1;
        if (wave == 0) {
            if (t + 1 < T) {
                const int nxt = cur ^ 1;
                tdm_load_2d((uint32_t)(uintptr_t)&As[nxt][0],
                            Atile + (size_t)(t + 1) * BK,
                            (uint32_t)K, (uint32_t)M, BK, BM, (uint64_t)K, 3u, 3u);
                tdm_load_2d((uint32_t)(uintptr_t)&Bs[nxt][0],
                            Btile + (size_t)(t + 1) * BK,
                            (uint32_t)K, (uint32_t)N, BK, BN, (uint64_t)K, 3u, 3u);
                __builtin_amdgcn_s_wait_tensorcnt(2);  // tile t landed (in-order)
            } else {
                __builtin_amdgcn_s_wait_tensorcnt(0);
            }
        }
        __syncthreads();   // tile t visible to all waves

        // ---- A fragments (ISA 16-bit A 16x32): dwords lhi*4+{0..3}, 8+lhi*4+{0..3}
        v16bf afr[2];
        #pragma unroll
        for (int mi = 0; mi < 2; ++mi) {
            const char* base = (const char*)&As[cur][0]
                             + (wm * 32 + mi * 16 + lrow) * (LDT * 2) + lhi * 16;
            u32x4* f = reinterpret_cast<u32x4*>(&afr[mi]);
            f[0] = *(const u32x4*)(base);
            f[1] = *(const u32x4*)(base + 32);
        }
        // ---- B fragments (ISA 16-bit B 32x16): dwords lhi*4+{0..3}, 8+lhi*4+{0..3}
        v16bf bfr[4];
        #pragma unroll
        for (int ni = 0; ni < 4; ++ni) {
            const char* base = (const char*)&Bs[cur][0]
                             + (wn * 64 + ni * 16 + lrow) * (LDT * 2) + lhi * 16;
            u32x4* f = reinterpret_cast<u32x4*>(&bfr[ni]);
            f[0] = *(const u32x4*)(base);
            f[1] = *(const u32x4*)(base + 32);
        }

        #pragma unroll
        for (int mi = 0; mi < 2; ++mi)
            #pragma unroll
            for (int ni = 0; ni < 4; ++ni)
                acc[mi][ni] = __builtin_amdgcn_wmma_f32_16x16x32_bf16(
                    false, afr[mi], false, bfr[ni],
                    (short)0, acc[mi][ni], false, false);

        __syncthreads();   // done reading buffers before TDM overwrites
    }

    // ---- fused epilogue: C = scale*relu(acc+bias) + extra ----
    const int rowBase = rowTile + wm * 32;
    const int colBase = colTile + wn * 64;
    #pragma unroll
    for (int ni = 0; ni < 4; ++ni) {
        const int col = colBase + ni * 16 + lrow;
        const float bv = bias[col];
        const float ev = (extra != nullptr) ? extra[col] : 0.0f;
        #pragma unroll
        for (int mi = 0; mi < 2; ++mi) {
            #pragma unroll
            for (int r = 0; r < 8; ++r) {
                const int row = rowBase + mi * 16 + lhi * 8 + r;
                float v = fmaxf(acc[mi][ni][r] + bv, 0.0f);
                C[(size_t)row * N + col] = f2bf(scale * v + ev);
            }
        }
    }
}

// one-shot f32 -> bf16 conversion (x)
__global__ __launch_bounds__(256)
void cvt_f32_bf16(const float* __restrict__ src, bf16* __restrict__ dst, int n)
{
    const int i = (blockIdx.x * blockDim.x + threadIdx.x) * 8;
    if (i + 8 <= n) {
        #pragma unroll
        for (int j = 0; j < 8; ++j) dst[i + j] = f2bf(src[i + j]);
    } else {
        for (int j = i; j < n; ++j) dst[j] = f2bf(src[j]);
    }
}

// one-shot f32 -> bf16 convert + transpose: W[K,N] -> Wt[N,K], 32x32 LDS tiles
__global__ __launch_bounds__(256)
void cvt_transpose_bf16(const float* __restrict__ W, bf16* __restrict__ Wt,
                        int K, int N)
{
    __shared__ bf16 tile[32][34];
    const int bn = blockIdx.x * 32;          // N offset
    const int bk = blockIdx.y * 32;          // K offset
    const int tx = threadIdx.x & 31;
    const int ty = threadIdx.x >> 5;         // 0..7
    #pragma unroll
    for (int r = 0; r < 32; r += 8)
        tile[ty + r][tx] = f2bf(W[(size_t)(bk + ty + r) * N + bn + tx]);
    __syncthreads();
    #pragma unroll
    for (int r = 0; r < 32; r += 8)
        Wt[(size_t)(bn + ty + r) * K + bk + tx] = tile[tx][ty + r];
}

// init_sum[j] = sum over the 3 nodes of init_nodes[n][j]
__global__ void init_sum_kernel(const float* __restrict__ init_nodes,
                                float* __restrict__ init_sum, int H)
{
    int j = blockIdx.x * blockDim.x + threadIdx.x;
    if (j < H) init_sum[j] = init_nodes[j] + init_nodes[H + j] + init_nodes[2 * H + j];
}

// out[b, 0..2] = dot(h2[b,:], dec_w) + dec_b   (all 3 nodes identical)
__global__ __launch_bounds__(256)
void decoder_kernel(const bf16* __restrict__ H2, const float* __restrict__ dec_w,
                    const float* __restrict__ dec_b, float* __restrict__ out,
                    int B, int H)
{
    const int lane = threadIdx.x & 31;
    const int row  = blockIdx.x * (blockDim.x >> 5) + (threadIdx.x >> 5);
    if (row >= B) return;
    const bf16* h = H2 + (size_t)row * H;
    float s = 0.0f;
    for (int j = lane; j < H; j += 32) s += bf2f(h[j]) * dec_w[j];
    #pragma unroll
    for (int off = 16; off > 0; off >>= 1) s += __shfl_down(s, off, 32);
    if (lane == 0) {
        const float v = s + dec_b[0];
        out[row * 3 + 0] = v;
        out[row * 3 + 1] = v;
        out[row * 3 + 2] = v;
    }
}

extern "C" void kernel_launch(void* const* d_in, const int* in_sizes, int n_in,
                              void* d_out, int out_size, void* d_ws, size_t ws_size,
                              hipStream_t stream)
{
    (void)in_sizes; (void)n_in; (void)out_size; (void)ws_size;
    constexpr int B   = 32768;
    constexpr int IN  = 512;
    constexpr int HID = 1024;

    const float* x        = (const float*)d_in[0];
    const float* enc_w1   = (const float*)d_in[1];
    const float* enc_b1   = (const float*)d_in[2];
    const float* enc_w2   = (const float*)d_in[3];
    const float* enc_b2   = (const float*)d_in[4];
    const float* init_nd  = (const float*)d_in[5];
    const float* gin1_w1  = (const float*)d_in[6];
    const float* gin1_b1  = (const float*)d_in[7];
    const float* gin1_w2  = (const float*)d_in[8];
    const float* gin1_b2  = (const float*)d_in[9];
    const float* gin2_w1  = (const float*)d_in[10];
    const float* gin2_b1  = (const float*)d_in[11];
    const float* gin2_w2  = (const float*)d_in[12];
    const float* gin2_b2  = (const float*)d_in[13];
    const float* dec_w    = (const float*)d_in[14];
    const float* dec_b    = (const float*)d_in[15];
    float* out = (float*)d_out;

    // workspace: bf16 x + transposed bf16 weights + ping-pong activations
    bf16* xbf  = (bf16*)d_ws;
    bf16* w1t  = xbf  + (size_t)B * IN;          // [HID, IN]
    bf16* w2t  = w1t  + (size_t)IN * HID;        // [HID, HID]
    bf16* g11t = w2t  + (size_t)HID * HID;
    bf16* g12t = g11t + (size_t)HID * HID;
    bf16* g21t = g12t + (size_t)HID * HID;
    bf16* g22t = g21t + (size_t)HID * HID;
    bf16* bufA = g22t + (size_t)HID * HID;
    bf16* bufB = bufA + (size_t)B * HID;
    float* init_sum = (float*)(bufB + (size_t)B * HID);

    const dim3 cblk(256);
    cvt_f32_bf16<<<dim3((B * IN / 8 + 255) / 256), cblk, 0, stream>>>(x, xbf, B * IN);
    const dim3 tg1(HID / 32, IN / 32);    // W[IN,HID] -> Wt[HID,IN]
    const dim3 tgH(HID / 32, HID / 32);
    cvt_transpose_bf16<<<tg1, cblk, 0, stream>>>(enc_w1,  w1t,  IN,  HID);
    cvt_transpose_bf16<<<tgH, cblk, 0, stream>>>(enc_w2,  w2t,  HID, HID);
    cvt_transpose_bf16<<<tgH, cblk, 0, stream>>>(gin1_w1, g11t, HID, HID);
    cvt_transpose_bf16<<<tgH, cblk, 0, stream>>>(gin1_w2, g12t, HID, HID);
    cvt_transpose_bf16<<<tgH, cblk, 0, stream>>>(gin2_w1, g21t, HID, HID);
    cvt_transpose_bf16<<<tgH, cblk, 0, stream>>>(gin2_w2, g22t, HID, HID);
    init_sum_kernel<<<dim3((HID + 255) / 256), cblk, 0, stream>>>(init_nd, init_sum, HID);

    const dim3 blk(256);
    const dim3 grd(HID / BN, B / BM);

    // 1) e1 = relu(x @ W1 + b1)
    gemm_bf16_wmma<<<grd, blk, 0, stream>>>(xbf,  w1t,  enc_b1, nullptr,
                                            1.0f, bufA, B, HID, IN);
    // 2) s = init_sum + 3*relu(e1 @ W2 + b2)
    gemm_bf16_wmma<<<grd, blk, 0, stream>>>(bufA, w2t,  enc_b2, init_sum,
                                            3.0f, bufB, B, HID, HID);
    // 3) m1 = relu(s @ g1w1 + g1b1)
    gemm_bf16_wmma<<<grd, blk, 0, stream>>>(bufB, g11t, gin1_b1, nullptr,
                                            1.0f, bufA, B, HID, HID);
    // 4) t = 3*relu(m1 @ g1w2 + g1b2)   (3x = second triangle aggregation)
    gemm_bf16_wmma<<<grd, blk, 0, stream>>>(bufA, g12t, gin1_b2, nullptr,
                                            3.0f, bufB, B, HID, HID);
    // 5) m2 = relu(t @ g2w1 + g2b1)
    gemm_bf16_wmma<<<grd, blk, 0, stream>>>(bufB, g21t, gin2_b1, nullptr,
                                            1.0f, bufA, B, HID, HID);
    // 6) h2 = relu(m2 @ g2w2 + g2b2)
    gemm_bf16_wmma<<<grd, blk, 0, stream>>>(bufA, g22t, gin2_b2, nullptr,
                                            1.0f, bufB, B, HID, HID);
    // 7) out[b, 0..2] = h2 . dec_w + dec_b
    decoder_kernel<<<dim3(B / 8), dim3(256), 0, stream>>>(bufB, dec_w, dec_b, out, B, HID);
}
